// HydraAttention_2585570312780
// MI455X (gfx1250) — matrix operand order repaired and checked by reading, not compiled, not run on hardware
//
#include <hip/hip_runtime.h>

// Hydra attention fused: qkv = x@W + b; out = (q/|q|_T) * sum_T((k/|k|_T)*v)
// x (4,4096,2048) f32, W (2048,6144) f32, b (6144) f32, out (4,4096,2048) f32.
// bf16 WMMA GEMM (f32 accum). q -> d_out; k,v folded into per-(batch,channel)
// reductions Sqq/Skk/Skv in d_ws (96 KB). Final pass scales q in place.
//
// Block: 1024 threads (32 waves), tile 256x128 per q/k/v, wave tile 32x32.
// Double-buffered LDS pipeline: next x tile staged via async global->LDS
// (ASYNCcnt) and next W tile loaded into VGPRs while current tile's 12 WMMAs
// run; W regs are cvt_pk'd and ds_stored to the back buffer after compute.

#define D_MODEL   2048
#define THREE_D   6144
#define M_TOTAL   16384          // B*T
#define TILE_M    256
#define TILE_N    128
#define BK        32
#define A_STRIDE  36             // floats per LDS row (32 + pad, keeps 16B align)
#define B_STRIDE  20             // dwords per LDS row (16 + pad, keeps 16B align)

typedef __attribute__((ext_vector_type(16))) __bf16 v16bf;
typedef __attribute__((ext_vector_type(2)))  __bf16 v2bf;
typedef __attribute__((ext_vector_type(8)))  float  v8f;
typedef __attribute__((ext_vector_type(4)))  int    v4i;

typedef __attribute__((address_space(1))) v4i* gas_v4i_ptr;
typedef __attribute__((address_space(3))) v4i* las_v4i_ptr;

union FragU {
    unsigned u[8];
    v16bf    v;
};

#if defined(__AMDGCN__) && __has_builtin(__builtin_amdgcn_global_load_async_to_lds_b128) && __has_builtin(__builtin_amdgcn_s_wait_asynccnt)
#define USE_ASYNC 1
#else
#define USE_ASYNC 0
#endif

static __device__ __forceinline__ unsigned pk_bf16(float lo, float hi) {
#if defined(__AMDGCN__) && __has_builtin(__builtin_amdgcn_cvt_pk_bf16_f32)
    v2bf r = __builtin_amdgcn_cvt_pk_bf16_f32(lo, hi);
    return __builtin_bit_cast(unsigned, r);
#else
    unsigned ulo = __builtin_bit_cast(unsigned, lo);
    unsigned uhi = __builtin_bit_cast(unsigned, hi);
    ulo = (ulo + 0x7FFFu + ((ulo >> 16) & 1u)) >> 16;
    uhi = (uhi + 0x7FFFu + ((uhi >> 16) & 1u)) & 0xFFFF0000u;
    return uhi | ulo;
#endif
}

__global__ __launch_bounds__(256) void hydra_zero_red(float* __restrict__ red) {
    int i = blockIdx.x * 256 + threadIdx.x;
    if (i < 3 * 4 * D_MODEL) red[i] = 0.0f;
}

__global__ __launch_bounds__(1024) void hydra_qkv_wmma(
        const float* __restrict__ x, const float* __restrict__ W,
        const float* __restrict__ bias, float* __restrict__ out,
        float* __restrict__ red) {
    __shared__ __align__(16) float    A_f32[2][TILE_M * A_STRIDE];      // 72 KB
    __shared__ __align__(16) unsigned B_bf16[2][3 * TILE_N * B_STRIDE]; // 60 KB

    const int tid  = threadIdx.x;
    const int n0   = blockIdx.x * TILE_N;
    const int m0   = blockIdx.y * TILE_M;
    const int wave = tid >> 5;
    const int lane = tid & 31;
    const int lm   = lane & 15;                  // M (A) / N (B, C)
    const int kh   = lane >> 4;                  // K-half selector
    const int wr   = wave >> 2;                  // 0..7: 32-row strip
    const int wc   = wave & 3;                   // 0..3: 32-col strip

    // staging indices (constant per thread)
    const int am = tid >> 3, ac = tid & 7;       // A: rows 0..127 (+128 on i=1)
    const int bn = tid & 127, bkw = tid >> 7;    // B: n, kw (kw+8 on i=1)

    v8f acc[3][2][2];
    #pragma unroll
    for (int s = 0; s < 3; ++s)
        #pragma unroll
        for (int nt = 0; nt < 2; ++nt)
            #pragma unroll
            for (int ar = 0; ar < 2; ++ar) acc[s][nt][ar] = {};

    float br0[6], br1[6];                        // pipelined W values

    // ---- helpers as lambdas ----
    auto stageA = [&](int buf, int kk) {
        #pragma unroll
        for (int i = 0; i < 2; ++i) {
            int m = am + i * 128;
            const float* src = x + (size_t)(m0 + m) * D_MODEL + kk + ac * 4;
            float* dst = &A_f32[buf][m * A_STRIDE + ac * 4];
#if USE_ASYNC
            __builtin_amdgcn_global_load_async_to_lds_b128(
                (gas_v4i_ptr)(v4i*)const_cast<float*>(src),
                (las_v4i_ptr)(v4i*)dst, 0, 0);
#else
            *(float4*)dst = *(const float4*)src;
#endif
        }
    };
    auto loadB = [&](int kk) {
        #pragma unroll
        for (int s = 0; s < 3; ++s)
            #pragma unroll
            for (int i = 0; i < 2; ++i) {
                int kw = bkw + i * 8;
                const float* p = W + (kk + 2 * kw) * THREE_D + s * D_MODEL + n0 + bn;
                br0[s * 2 + i] = p[0];
                br1[s * 2 + i] = p[THREE_D];
            }
    };
    auto storeB = [&](int buf) {
        #pragma unroll
        for (int s = 0; s < 3; ++s)
            #pragma unroll
            for (int i = 0; i < 2; ++i) {
                int kw = bkw + i * 8;
                B_bf16[buf][(s * TILE_N + bn) * B_STRIDE + kw] =
                    pk_bf16(br0[s * 2 + i], br1[s * 2 + i]);
            }
    };

    // ---- prologue: fill buffer 0 ----
    stageA(0, 0);
    loadB(0);
    storeB(0);

    for (int kk = 0; kk < D_MODEL; kk += BK) {
        const int cur = (kk >> 5) & 1;
        const bool more = (kk + BK) < D_MODEL;
#if USE_ASYNC
        __builtin_amdgcn_s_wait_asynccnt(0);
#endif
        __syncthreads();

        // launch next tile's traffic before computing on the current one
        if (more) {
            stageA(cur ^ 1, kk + BK);
            loadB(kk + BK);
        }

        // ---- A fragments (16x32 bf16, ISA layout), cvt f32->bf16 at load ----
        FragU a[2];
        #pragma unroll
        for (int ar = 0; ar < 2; ++ar) {
            const float* base =
                &A_f32[cur][(wr * 32 + ar * 16 + lm) * A_STRIDE + 8 * kh];
            #pragma unroll
            for (int jg = 0; jg < 2; ++jg) {
                float4 f0 = *(const float4*)(base + jg * 16);
                float4 f1 = *(const float4*)(base + jg * 16 + 4);
                a[ar].u[jg * 4 + 0] = pk_bf16(f0.x, f0.y);
                a[ar].u[jg * 4 + 1] = pk_bf16(f0.z, f0.w);
                a[ar].u[jg * 4 + 2] = pk_bf16(f1.x, f1.y);
                a[ar].u[jg * 4 + 3] = pk_bf16(f1.z, f1.w);
            }
        }
        // ---- 12 WMMAs: 3 matrices x 2 col tiles x 2 row tiles ----
        #pragma unroll
        for (int s = 0; s < 3; ++s) {
            #pragma unroll
            for (int nt = 0; nt < 2; ++nt) {
                FragU b;
                const uint4* bp = (const uint4*)
                    &B_bf16[cur][(s * TILE_N + wc * 32 + nt * 16 + lm) * B_STRIDE
                                 + 4 * kh];
                uint4 w0 = bp[0], w1 = bp[2];
                b.u[0] = w0.x; b.u[1] = w0.y; b.u[2] = w0.z; b.u[3] = w0.w;
                b.u[4] = w1.x; b.u[5] = w1.y; b.u[6] = w1.z; b.u[7] = w1.w;
                #pragma unroll
                for (int ar = 0; ar < 2; ++ar)
                    acc[s][nt][ar] = __builtin_amdgcn_wmma_f32_16x16x32_bf16(
                        false, a[ar].v, false, b.v, (short)0, acc[s][nt][ar],
                        false, false);
            }
        }

        // convert + store pipelined W regs into the back buffer (after compute,
        // so their global-load latency hides under the WMMAs)
        if (more) storeB(cur ^ 1);
    }

    // ---- epilogue: bias, write q, reduce q^2 / k^2 / k*v over 32-row strip ----
    const int batch = m0 >> 12;                  // m0 / 4096
    #pragma unroll
    for (int nt = 0; nt < 2; ++nt) {
        const int colg = n0 + wc * 32 + nt * 16 + lm;
        const float bq = bias[colg];
        const float bk = bias[D_MODEL + colg];
        const float bv = bias[2 * D_MODEL + colg];
        float sq = 0.0f, sk = 0.0f, skv = 0.0f;
        #pragma unroll
        for (int ar = 0; ar < 2; ++ar) {
            #pragma unroll
            for (int r = 0; r < 8; ++r) {
                const int rowg = m0 + wr * 32 + ar * 16 + r + 8 * kh;  // C/D layout
                const float qv = acc[0][nt][ar][r] + bq;
                const float kv = acc[1][nt][ar][r] + bk;
                const float vv = acc[2][nt][ar][r] + bv;
                out[(size_t)rowg * D_MODEL + colg] = qv;
                sq  += qv * qv;
                sk  += kv * kv;
                skv += kv * vv;
            }
        }
        sq  += __shfl_xor(sq, 16, 32);
        sk  += __shfl_xor(sk, 16, 32);
        skv += __shfl_xor(skv, 16, 32);
        if (kh == 0) {
            const int idx = batch * D_MODEL + colg;
            atomicAdd(&red[idx], sq);
            atomicAdd(&red[4 * D_MODEL + idx], sk);
            atomicAdd(&red[8 * D_MODEL + idx], skv);
        }
    }
}

__global__ __launch_bounds__(256) void hydra_scale(
        float* __restrict__ out, const float* __restrict__ red) {
    const size_t gid = (size_t)blockIdx.x * 256 + threadIdx.x;  // one float4 each
    const size_t e = gid * 4;
    const int row   = (int)(e >> 11);            // /2048
    const int colb  = (int)(e & 2047);
    const int batch = row >> 12;
    float4 q = *(float4*)(out + e);
    const float* sqq = red + batch * D_MODEL + colb;
    const float* skk = sqq + 4 * D_MODEL;
    const float* skv = sqq + 8 * D_MODEL;
    q.x *= skv[0] * rsqrtf(sqq[0] * skk[0]);
    q.y *= skv[1] * rsqrtf(sqq[1] * skk[1]);
    q.z *= skv[2] * rsqrtf(sqq[2] * skk[2]);
    q.w *= skv[3] * rsqrtf(sqq[3] * skk[3]);
    *(float4*)(out + e) = q;
}

extern "C" void kernel_launch(void* const* d_in, const int* in_sizes, int n_in,
                              void* d_out, int out_size, void* d_ws, size_t ws_size,
                              hipStream_t stream) {
    const float* x = (const float*)d_in[0];
    const float* W = (const float*)d_in[1];
    const float* b = (const float*)d_in[2];
    float* out = (float*)d_out;
    float* red = (float*)d_ws;   // needs 3*4*2048*4 = 96 KB

    hydra_zero_red<<<(3 * 4 * D_MODEL + 255) / 256, 256, 0, stream>>>(red);

    dim3 grid(D_MODEL / TILE_N, M_TOTAL / TILE_M);   // (16, 64)
    hydra_qkv_wmma<<<grid, 1024, 0, stream>>>(x, W, b, out, red);

    const int total4 = M_TOTAL * D_MODEL / 4;        // 8388608 float4s
    hydra_scale<<<total4 / 256, 256, 0, stream>>>(out, red);
}
